// GridLSTM_47399259079400
// MI455X (gfx1250) — matrix operand order repaired and checked by reading, not compiled, not run on hardware
//
#include <hip/hip_runtime.h>

#define HID 512
#define BATCH 64

typedef __attribute__((ext_vector_type(16))) _Float16 v16h;
typedef __attribute__((ext_vector_type(8)))  float    v8f;

// packed cell activation: [4 mt][16 kt][512] halves = 32768 halves / cell
#define CELLH 32768
// parity slab for packed h states: 32 cells
#define PARH ((size_t)32 * CELLH)
// parity slab for f32 c states
#define PARF ((size_t)32 * 64 * 512)

__device__ __forceinline__ v8f wmma16(v16h a, v16h b, v8f c) {
  return __builtin_amdgcn_wmma_f32_16x16x32_f16(
      false, a, false, b, (short)0, c, false, false);
}

__device__ __forceinline__ float sigf(float x) {
  return 1.0f / (1.0f + __expf(-x));
}

// A-fragment k mapping (ISA 7.12.2, 16-bit 16x32 A tile):
//   k_in_tile = (lane>>4)*8 + (j<8 ? j : j+8), m = lane&15
__device__ __forceinline__ int a_k_of(int lane, int j) {
  return ((lane >> 4) << 3) + (j < 8 ? j : j + 8);
}

// For element (m_local 0..15, k_in_tile 0..31) -> packed (lane, j) slot.
__device__ __forceinline__ int a_slot_of(int mloc, int kin) {
  int khalf = kin >> 4;
  int kk = kin & 15;
  int khi = kk >> 3;
  int jj = (kk & 7) + 8 * khalf;
  int lane_a = mloc | (khi << 4);
  return lane_a * 16 + jj;
}

// ---------------------------------------------------------------------------
// Pack row-major f32 weight [K, N] -> WMMA B-fragment f16.
// Tile t = nt*(K/32)+kt; element (lane, j): k = kt*32+(lane>>4)*8+(j<8?j:j+8),
// n = nt*16+(lane&15). Lane's 16 halves contiguous -> 2x global_load_b128.
// ---------------------------------------------------------------------------
__global__ void pack_weight(const float* __restrict__ src,
                            _Float16* __restrict__ dst, int K, int N) {
  int ktiles = K >> 5;
  int total = K * N;
  for (int idx = blockIdx.x * blockDim.x + threadIdx.x; idx < total;
       idx += gridDim.x * blockDim.x) {
    int t = idx >> 9;
    int e = idx & 511;
    int lane = e >> 4, j = e & 15;
    int nt = t / ktiles, kt = t % ktiles;
    int k = kt * 32 + a_k_of(lane, j);
    int n = nt * 16 + (lane & 15);
    dst[idx] = (_Float16)src[(size_t)k * N + n];
  }
}

// Pack f32 activations [32 cells][64, 512] -> A-fragment f16 per cell.
__global__ void pack_act(const float* __restrict__ src,
                         _Float16* __restrict__ dst) {
  int total = 32 * CELLH;
  for (int idx = blockIdx.x * blockDim.x + threadIdx.x; idx < total;
       idx += gridDim.x * blockDim.x) {
    int cell = idx >> 15;
    int e = idx & (CELLH - 1);
    int t = e >> 9;              // mt*16 + ktb
    int lane = (e >> 4) & 31, j = e & 15;
    int mt = t >> 4, ktb = t & 15;
    int m = mt * 16 + (lane & 15);
    int k = ktb * 32 + a_k_of(lane, j);
    dst[idx] = (_Float16)src[((size_t)cell * 64 + m) * 512 + k];
  }
}

__global__ void zero_f32(float* __restrict__ p, int n) {
  for (int i = blockIdx.x * blockDim.x + threadIdx.x; i < n;
       i += gridDim.x * blockDim.x)
    p[i] = 0.0f;
}

// ---------------------------------------------------------------------------
// Phase A: hmid[64,512] = [S_h | T_h] @ W_H2h + b1. Pure register GEMM:
// all A/B fragments are contiguous global v16h loads; no LDS, no barriers.
// Output written A-fragment-packed f16 for phase B.
// ---------------------------------------------------------------------------
__global__ __launch_bounds__(256) void phaseA(
    const _Float16* __restrict__ ShPk, const _Float16* __restrict__ ThPk,
    const _Float16* __restrict__ pkW1, const float* __restrict__ b1,
    _Float16* __restrict__ hmidpk, int d, int jlo) {
  int j = jlo + blockIdx.y;
  int i = d - j;
  int pr = (d + 1) & 1;
  const _Float16* Sc = ShPk + pr * PARH + (size_t)i * CELLH;
  const _Float16* Tc = ThPk + pr * PARH + (size_t)j * CELLH;

  int tid = threadIdx.x, lane = tid & 31, wv = tid >> 5;
  int ntile = blockIdx.x * 4 + (wv & 3);  // 16-col output tile, 0..31
  int mts = (wv >> 2) * 2;                // mt = mts, mts+1

  v8f acc0 = {0, 0, 0, 0, 0, 0, 0, 0};
  v8f acc1 = {0, 0, 0, 0, 0, 0, 0, 0};

  const _Float16* bw = pkW1 + ((size_t)ntile * 32) * 512 + lane * 16;
  const _Float16* a0s = Sc + (size_t)(mts * 16) * 512 + lane * 16;
  const _Float16* a1s = Sc + (size_t)((mts + 1) * 16) * 512 + lane * 16;
  const _Float16* a0t = Tc + (size_t)(mts * 16) * 512 + lane * 16;
  const _Float16* a1t = Tc + (size_t)((mts + 1) * 16) * 512 + lane * 16;

#pragma unroll 2
  for (int kt = 0; kt < 16; ++kt) {  // K tiles 0..15 from S_h
    v16h bf = *(const v16h*)(bw + (size_t)kt * 512);
    v16h a0 = *(const v16h*)(a0s + (size_t)kt * 512);
    v16h a1 = *(const v16h*)(a1s + (size_t)kt * 512);
    acc0 = wmma16(a0, bf, acc0);
    acc1 = wmma16(a1, bf, acc1);
  }
#pragma unroll 2
  for (int kt = 0; kt < 16; ++kt) {  // K tiles 16..31 from T_h
    v16h bf = *(const v16h*)(bw + (size_t)(16 + kt) * 512);
    v16h a0 = *(const v16h*)(a0t + (size_t)kt * 512);
    v16h a1 = *(const v16h*)(a1t + (size_t)kt * 512);
    acc0 = wmma16(a0, bf, acc0);
    acc1 = wmma16(a1, bf, acc1);
  }

  // bias + scatter-store as packed A fragments of hmid
  float bias = b1[ntile * 16 + (lane & 15)];
  int khiL = lane >> 4;
  int col = ntile * 16 + (lane & 15);
  int ktb = col >> 5, kin = col & 31;
  _Float16* hout = hmidpk + (size_t)j * CELLH;
#pragma unroll
  for (int q = 0; q < 2; ++q) {
    int mt = mts + q;
    v8f a = q ? acc1 : acc0;
#pragma unroll
    for (int r = 0; r < 8; ++r) {
      int mloc = r + 8 * khiL;
      hout[(size_t)(mt * 16 + ktb) * 512 + a_slot_of(mloc, kin)] =
          (_Float16)(a[r] + bias);
    }
  }
}

// ---------------------------------------------------------------------------
// Phase B+C fused: z_s = x_s@Wk + hmid@Wr + b, z_t = x_t@Wk + hmid@Wr + b
// (shared hmid@Wr accumulated once), gate exchange via LDS, LSTM gate math,
// packed-f16 h writeback + f32 c writeback + final outputs.
// ---------------------------------------------------------------------------
__global__ __launch_bounds__(256) void phaseBC(
    const _Float16* __restrict__ pkSrc, const _Float16* __restrict__ pkTgt,
    _Float16* __restrict__ ShPk, _Float16* __restrict__ ThPk,
    float* __restrict__ Sm, float* __restrict__ Tm,
    const _Float16* __restrict__ pkWk, const _Float16* __restrict__ pkWr,
    const _Float16* __restrict__ hmidpk, const float* __restrict__ bl,
    float* __restrict__ out, int d, int jlo) {
  __shared__ float ldsZ[2][4][64][16];  // z_s/z_t x gate x row x col (32 KB)

  int j = jlo + blockIdx.y;
  int i = d - j;
  int pr = (d + 1) & 1, pw = d & 1;
  const _Float16* xs = (j == 0) ? (pkSrc + (size_t)i * CELLH)
                                : (ShPk + pr * PARH + (size_t)i * CELLH);
  const _Float16* xt = (i == 0) ? (pkTgt + (size_t)j * CELLH)
                                : (ThPk + pr * PARH + (size_t)j * CELLH);
  const _Float16* hm = hmidpk + (size_t)j * CELLH;

  int tid = threadIdx.x, lane = tid & 31, wv = tid >> 5;
  int g = wv & 3;           // gate handled by this wave
  int mts = (wv >> 2) * 2;  // mt = mts, mts+1
  int cg = blockIdx.x;      // 16-col group within H
  int ztile = g * 32 + cg;  // 16-col tile in the 2048-wide z

  v8f zv = {0, 0, 0, 0, 0, 0, 0, 0};
  v8f aSH[2] = {zv, zv}, aS[2] = {zv, zv}, aT[2] = {zv, zv};

  const _Float16* bwr = pkWr + ((size_t)ztile * 16) * 512 + lane * 16;
  const _Float16* bwk = pkWk + ((size_t)ztile * 16) * 512 + lane * 16;

#pragma unroll 2
  for (int kt = 0; kt < 16; ++kt) {  // K = 512 -> 16 tiles
    v16h br = *(const v16h*)(bwr + (size_t)kt * 512);
    v16h bk = *(const v16h*)(bwk + (size_t)kt * 512);
#pragma unroll
    for (int q = 0; q < 2; ++q) {
      size_t fo = (size_t)((mts + q) * 16 + kt) * 512 + lane * 16;
      v16h ah = *(const v16h*)(hm + fo);
      v16h as = *(const v16h*)(xs + fo);
      v16h at = *(const v16h*)(xt + fo);
      aSH[q] = wmma16(ah, br, aSH[q]);
      aS[q] = wmma16(as, bk, aS[q]);
      aT[q] = wmma16(at, bk, aT[q]);
    }
  }

  // z = shared + own + bias -> LDS exchange across gates
  float bz = bl[g * 512 + cg * 16 + (lane & 15)];
  int khiL = lane >> 4;
  int n0 = lane & 15;
#pragma unroll
  for (int q = 0; q < 2; ++q) {
    int mt = mts + q;
#pragma unroll
    for (int r = 0; r < 8; ++r) {
      int m = mt * 16 + r + 8 * khiL;
      ldsZ[0][g][m][n0] = aSH[q][r] + aS[q][r] + bz;
      ldsZ[1][g][m][n0] = aSH[q][r] + aT[q][r] + bz;
    }
  }
  __syncthreads();

  // Fused gate math for 64 rows x 16 cols owned by this block.
  const float* SmR = Sm + pr * PARF + (size_t)i * 64 * 512;
  const float* TmR = Tm + pr * PARF + (size_t)j * 64 * 512;
  float* SmW = Sm + pw * PARF + (size_t)i * 64 * 512;
  float* TmW = Tm + pw * PARF + (size_t)j * 64 * 512;
  _Float16* ShW = ShPk + pw * PARH + (size_t)i * CELLH;
  _Float16* ThW = ThPk + pw * PARH + (size_t)j * CELLH;

  for (int p = tid; p < 1024; p += 256) {
    int m = p >> 4, n = p & 15;
    int col = cg * 16 + n;
    int mt = m >> 4, mloc = m & 15;
    int ktb = col >> 5, kin = col & 31;
    size_t pslot = (size_t)(mt * 16 + ktb) * 512 + a_slot_of(mloc, kin);
    // source cell
    float zi = ldsZ[0][0][m][n], zf = ldsZ[0][1][m][n];
    float zg = ldsZ[0][2][m][n], zo = ldsZ[0][3][m][n];
    float c = SmR[m * 512 + col];
    float c2 = sigf(zf) * c + sigf(zi) * tanhf(zg);
    float h2 = sigf(zo) * tanhf(c2);
    SmW[m * 512 + col] = c2;
    ShW[pslot] = (_Float16)h2;
    if (j == 31) out[((size_t)i * 64 + m) * 512 + col] = h2;
    // target cell
    zi = ldsZ[1][0][m][n]; zf = ldsZ[1][1][m][n];
    zg = ldsZ[1][2][m][n]; zo = ldsZ[1][3][m][n];
    c = TmR[m * 512 + col];
    c2 = sigf(zf) * c + sigf(zi) * tanhf(zg);
    h2 = sigf(zo) * tanhf(c2);
    TmW[m * 512 + col] = c2;
    ThW[pslot] = (_Float16)h2;
    if (i == 31)
      out[(size_t)32 * 64 * 512 + ((size_t)j * 64 + m) * 512 + col] = h2;
  }
}

extern "C" void kernel_launch(void* const* d_in, const int* in_sizes, int n_in,
                              void* d_out, int out_size, void* d_ws,
                              size_t ws_size, hipStream_t stream) {
  (void)in_sizes; (void)n_in; (void)out_size; (void)ws_size;
  const float* src = (const float*)d_in[0];  // [32,64,512]
  const float* tgt = (const float*)d_in[1];  // [32,64,512]
  const float* W1 = (const float*)d_in[2];   // [1024,512]
  const float* b1 = (const float*)d_in[3];   // [512]
  const float* Wk = (const float*)d_in[4];   // [512,2048]
  const float* Wr = (const float*)d_in[5];   // [512,2048]
  const float* bl = (const float*)d_in[6];   // [2048]
  float* out = (float*)d_out;

  // workspace layout (~35 MB)
  _Float16* pkW1 = (_Float16*)d_ws;          // 1024*512
  _Float16* pkWk = pkW1 + 524288;            // 512*2048
  _Float16* pkWr = pkWk + 1048576;           // 512*2048
  _Float16* hmid = pkWr + 1048576;           // 32 * CELLH
  _Float16* pkSrc = hmid + 32 * CELLH;       // 32 * CELLH
  _Float16* pkTgt = pkSrc + 32 * CELLH;      // 32 * CELLH
  _Float16* ShPk = pkTgt + 32 * CELLH;       // [2][32]*CELLH
  _Float16* ThPk = ShPk + 2 * PARH;          // [2][32]*CELLH
  float* Sm = (float*)(ThPk + 2 * PARH);     // [2][32][64][512]
  float* Tm = Sm + 2 * PARF;                 // [2][32][64][512]

  pack_weight<<<dim3(512), dim3(256), 0, stream>>>(W1, pkW1, 1024, 512);
  pack_weight<<<dim3(1024), dim3(256), 0, stream>>>(Wk, pkWk, 512, 2048);
  pack_weight<<<dim3(1024), dim3(256), 0, stream>>>(Wr, pkWr, 512, 2048);
  pack_act<<<dim3(1024), dim3(256), 0, stream>>>(src, pkSrc);
  pack_act<<<dim3(1024), dim3(256), 0, stream>>>(tgt, pkTgt);
  // zero packed h states (f16 zeros == zero bits) and f32 c states
  zero_f32<<<dim3(2048), dim3(256), 0, stream>>>((float*)ShPk,
                                                 (int)(2 * PARH));  // 4M h = 2M f32 x2 arrays
  zero_f32<<<dim3(2048), dim3(256), 0, stream>>>(Sm, (int)(4 * PARF));

  // wavefront over anti-diagonals d = i + j
  for (int dg = 0; dg <= 62; ++dg) {
    int jlo = dg > 31 ? dg - 31 : 0;
    int jhi = dg < 31 ? dg : 31;
    int nc = jhi - jlo + 1;
    phaseA<<<dim3(8, nc), dim3(256), 0, stream>>>(ShPk, ThPk, pkW1, b1, hmid,
                                                  dg, jlo);
    phaseBC<<<dim3(32, nc), dim3(256), 0, stream>>>(
        pkSrc, pkTgt, ShPk, ThPk, Sm, Tm, pkWk, pkWr, hmid, bl, out, dg, jlo);
  }
}
